// Attention_89489938579587
// MI455X (gfx1250) — matrix operand order repaired and checked by reading, not compiled, not run on hardware
//
#include <hip/hip_runtime.h>
#include <cmath>

// ---------------------------------------------------------------------------
// MI455X / gfx1250 attention layer, fp32 I/O.
// Matmuls: fp32 emulated via split-bf16 WMMA (AhiBhi + AhiBlo + AloBhi).
// q/k/v intermediates stored as pre-split bf16 hi/lo planes so the flash
// kernel's LDS fills are pure async copies (global_load_async_to_lds_b128,
// double-buffered, ASYNCcnt-synchronized).
//
// Workspace layout (bytes):
//   qkh : [8192][4096] bf16   (q|k hi plane)            67.1 MB
//   qkl : [8192][4096] bf16   (q|k lo plane)            67.1 MB
//   vTh : [B*H][HD][T] bf16   (V transposed, hi)        33.6 MB
//   vTl : [B*H][HD][T] bf16   (V transposed, lo)        33.6 MB
//   y   : [8192][2048] f32    (attention output)        67.1 MB
// ---------------------------------------------------------------------------

typedef __attribute__((ext_vector_type(16))) __bf16 v16bf;
typedef __attribute__((ext_vector_type(8)))  __bf16 v8bf;
typedef __attribute__((ext_vector_type(4)))  __bf16 v4bf;
typedef __attribute__((ext_vector_type(8)))  float  v8f;
typedef __attribute__((ext_vector_type(4)))  float  v4f;

#define B_    4
#define T_    2048
#define DIM_  2048
#define H_    16
#define HD_   128
#define QK_LD 4096          // row stride (elements) of the q|k planes

struct bfpair { __bf16 hi, lo; };

static __device__ __forceinline__ bfpair split1(float x) {
  bfpair r;
  r.hi = (__bf16)x;
  r.lo = (__bf16)(x - (float)r.hi);
  return r;
}

static __device__ __forceinline__ v16bf cat8(v8bf a, v8bf b) {
  v16bf r;
#pragma unroll
  for (int i = 0; i < 8; ++i) { r[i] = a[i]; r[i + 8] = b[i]; }
  return r;
}

static __device__ __forceinline__ v8f v8f_zero() {
  v8f r;
#pragma unroll
  for (int i = 0; i < 8; ++i) r[i] = 0.0f;
  return r;
}

static __device__ __forceinline__ v8f wmma_bf16(v16bf a, v16bf b, v8f c) {
  return __builtin_amdgcn_wmma_f32_16x16x32_bf16(false, a, false, b, (short)0, c,
                                                 false, false);
}

// A-fragment (16x32): lane<16 -> K{0..7,16..23}, lane>=16 -> K{8..15,24..31}
static __device__ __forceinline__ v16bf ldA(const __bf16* rowbase, int off0) {
  return cat8(*(const v8bf*)(rowbase + off0), *(const v8bf*)(rowbase + off0 + 16));
}
// B-fragment (32x16): lane<16 -> K{0..7,16..23}, lane>=16 -> K{8..15,24..31}
static __device__ __forceinline__ v16bf ldB(const __bf16* colbase, int lane) {
  int o = (lane < 16) ? 0 : 8;
  return cat8(*(const v8bf*)(colbase + o), *(const v8bf*)(colbase + o + 16));
}

static __device__ __forceinline__ float hmax16(float v) {
#pragma unroll
  for (int m = 1; m < 16; m <<= 1) v = fmaxf(v, __shfl_xor(v, m, 32));
  return v;
}
static __device__ __forceinline__ float hsum16(float v) {
#pragma unroll
  for (int m = 1; m < 16; m <<= 1) v += __shfl_xor(v, m, 32);
  return v;
}

// 16-byte async copy global -> LDS (ASYNCcnt tracked, per-lane addresses)
static __device__ __forceinline__ void async_cp16(const __bf16* lds,
                                                  const __bf16* gbase,
                                                  unsigned voff_bytes) {
  unsigned l = (unsigned)(unsigned long long)(const void*)lds;
  asm volatile("global_load_async_to_lds_b128 %0, %1, %2"
               :: "v"(l), "v"(voff_bytes), "s"(gbase) : "memory");
}
static __device__ __forceinline__ void wait_async0() {
  asm volatile("s_wait_asynccnt 0x0" ::: "memory");
}

// ---------------------------------------------------------------------------
// C[M,N] = A[M,K] * Bm[N,K]^T, split-bf16, block tile 128x128, K-step 32,
// 8 waves x (64x32). Register-staged prefetch of the next K slice overlaps
// global loads with the 24 WMMAs per step.
// mode 0: store C as f32.
// mode 1: qkv epilogue -> n<4096 to qk hi/lo planes; n>=4096 to transposed
//         V planes vT[(b*H+h)*HD + d][t] (contiguous along the C VGPR index).
// ---------------------------------------------------------------------------
#define LDT 72  // LDS row stride in bf16 (144B: 16B-aligned, conflict-light)

__global__ __launch_bounds__(256)
void gemm_nt_kernel(const float* __restrict__ A, const float* __restrict__ Bm,
                    float* __restrict__ C,
                    __bf16* __restrict__ qkh, __bf16* __restrict__ qkl,
                    __bf16* __restrict__ vTh, __bf16* __restrict__ vTl,
                    int M, int N, int K, int mode) {
  __shared__ __bf16 Ah[128 * LDT], Al[128 * LDT];
  __shared__ __bf16 Bh[128 * LDT], Bl[128 * LDT];

  const int tid   = threadIdx.x;
  const int lane  = tid & 31;
  const int wave  = tid >> 5;
  const int l16   = lane & 15;
  const int half8 = (lane < 16) ? 0 : 8;
  const int bm = blockIdx.y * 128;
  const int bn = blockIdx.x * 128;
  const int wm = (wave >> 2) * 64;
  const int wn = (wave & 3) * 32;

  v8f acc[4][2];
#pragma unroll
  for (int i = 0; i < 4; ++i)
#pragma unroll
    for (int j = 0; j < 2; ++j) acc[i][j] = v8f_zero();

  int frow[4], fc4[4];
#pragma unroll
  for (int r = 0; r < 4; ++r) {
    int slot = tid + r * 256;      // 1024 float4 slots = 128 rows x 8 col4
    frow[r] = slot >> 3;
    fc4[r]  = (slot & 7) * 4;
  }

  // prologue: prefetch K-slice 0 into registers
  v4f avr[4], bvr[4];
#pragma unroll
  for (int r = 0; r < 4; ++r) {
    avr[r] = *(const v4f*)(A  + (size_t)(bm + frow[r]) * K + fc4[r]);
    bvr[r] = *(const v4f*)(Bm + (size_t)(bn + frow[r]) * K + fc4[r]);
  }

  for (int k0 = 0; k0 < K; k0 += 32) {
    __syncthreads();
    // stage current slice (convert fp32 -> hi/lo bf16)
#pragma unroll
    for (int r = 0; r < 4; ++r) {
      v4bf ah, al, bh, bl;
#pragma unroll
      for (int j = 0; j < 4; ++j) {
        bfpair pa = split1(avr[r][j]); ah[j] = pa.hi; al[j] = pa.lo;
        bfpair pb = split1(bvr[r][j]); bh[j] = pb.hi; bl[j] = pb.lo;
      }
      *(v4bf*)&Ah[frow[r] * LDT + fc4[r]] = ah;
      *(v4bf*)&Al[frow[r] * LDT + fc4[r]] = al;
      *(v4bf*)&Bh[frow[r] * LDT + fc4[r]] = bh;
      *(v4bf*)&Bl[frow[r] * LDT + fc4[r]] = bl;
    }
    __syncthreads();
    // prefetch next slice while computing on this one
    if (k0 + 32 < K) {
#pragma unroll
      for (int r = 0; r < 4; ++r) {
        avr[r] = *(const v4f*)(A  + (size_t)(bm + frow[r]) * K + k0 + 32 + fc4[r]);
        bvr[r] = *(const v4f*)(Bm + (size_t)(bn + frow[r]) * K + k0 + 32 + fc4[r]);
      }
    }

    v16bf aHf[4], aLf[4];
#pragma unroll
    for (int ms = 0; ms < 4; ++ms) {
      aHf[ms] = ldA(&Ah[(wm + ms * 16 + l16) * LDT], half8);
      aLf[ms] = ldA(&Al[(wm + ms * 16 + l16) * LDT], half8);
    }
#pragma unroll
    for (int ns = 0; ns < 2; ++ns) {
      v16bf bH = ldB(&Bh[(wn + ns * 16 + l16) * LDT], lane);
      v16bf bL = ldB(&Bl[(wn + ns * 16 + l16) * LDT], lane);
#pragma unroll
      for (int ms = 0; ms < 4; ++ms) {
        acc[ms][ns] = wmma_bf16(aHf[ms], bH, acc[ms][ns]);
        acc[ms][ns] = wmma_bf16(aHf[ms], bL, acc[ms][ns]);
        acc[ms][ns] = wmma_bf16(aLf[ms], bH, acc[ms][ns]);
      }
    }
  }

  // epilogue. C layout: VGPR i -> m = base + half8 + i, n = base + l16
  if (mode == 0) {
#pragma unroll
    for (int ms = 0; ms < 4; ++ms)
#pragma unroll
      for (int ns = 0; ns < 2; ++ns)
#pragma unroll
        for (int i = 0; i < 8; ++i) {
          int m = bm + wm + ms * 16 + half8 + i;
          int n = bn + wn + ns * 16 + l16;
          C[(size_t)m * N + n] = acc[ms][ns][i];
        }
  } else {
#pragma unroll
    for (int ms = 0; ms < 4; ++ms)
#pragma unroll
      for (int ns = 0; ns < 2; ++ns) {
        int mbase = bm + wm + ms * 16 + half8;
        int n     = bn + wn + ns * 16 + l16;
        if (n < 2 * DIM_) {
          // q|k region: scalar b16 stores (stride QK_LD along the VGPR index)
#pragma unroll
          for (int i = 0; i < 8; ++i) {
            bfpair p = split1(acc[ms][ns][i]);
            size_t idx = (size_t)(mbase + i) * QK_LD + n;
            qkh[idx] = p.hi;
            qkl[idx] = p.lo;
          }
        } else {
          // V region -> transposed planes; contiguous along i (= t)
          int hc = n - 2 * DIM_;
          int h  = hc >> 7, d = hc & 127;
          int b  = mbase >> 11;          // m = b*T + t, tiles never straddle b
          int t0 = mbase & (T_ - 1);
          v8bf vh, vl;
#pragma unroll
          for (int i = 0; i < 8; ++i) {
            bfpair p = split1(acc[ms][ns][i]);
            vh[i] = p.hi; vl[i] = p.lo;
          }
          size_t idx = ((size_t)((b * H_ + h) * HD_ + d)) * T_ + t0;
          *(v8bf*)(vTh + idx) = vh;
          *(v8bf*)(vTl + idx) = vl;
        }
      }
  }
}

// ---------------------------------------------------------------------------
// RoPE in-place on the q|k hi/lo planes: recombine (exact), rotate, re-split.
// ---------------------------------------------------------------------------
__global__ __launch_bounds__(256)
void rope_kernel(__bf16* __restrict__ qkh, __bf16* __restrict__ qkl) {
  int idx = blockIdx.x * 256 + threadIdx.x;  // B*T*2*H*(HD/2) = 16,777,216
  int i   = idx & 63;
  int h   = (idx >> 6) & 15;
  int reg = (idx >> 10) & 1;
  int row = idx >> 11;          // b*T + t
  int t   = row & (T_ - 1);
  size_t o = (size_t)row * QK_LD + reg * DIM_ + h * HD_ + 2 * i;
  float invf = __expf(-9.210340371976184f * (float)i * (1.0f / 64.0f));
  float ang  = (float)t * invf;
  float sv, cv;
  sincosf(ang, &sv, &cv);
  float x0 = (float)qkh[o]     + (float)qkl[o];
  float x1 = (float)qkh[o + 1] + (float)qkl[o + 1];
  float r0 = x0 * cv - x1 * sv;
  float r1 = x1 * cv + x0 * sv;
  bfpair p0 = split1(r0), p1 = split1(r1);
  qkh[o]     = p0.hi; qkl[o]     = p0.lo;
  qkh[o + 1] = p1.hi; qkl[o + 1] = p1.lo;
}

// ---------------------------------------------------------------------------
// Flash attention: block = (b,h, 128 q rows), 8 waves x 16 q rows.
// K and V tiles are raw bf16 copies -> double-buffered async loads to LDS.
// ---------------------------------------------------------------------------
#define KLD 136  // K tile row stride (272B, 16B-aligned, conflict-free)
#define VLD 48   // V^T / P row stride (96B, 16B-aligned)

__global__ __launch_bounds__(256)
void flash_attn_kernel(const __bf16* __restrict__ qkh, const __bf16* __restrict__ qkl,
                       const __bf16* __restrict__ vTh, const __bf16* __restrict__ vTl,
                       float* __restrict__ y) {
  __shared__ __bf16 KhB[2 * 32 * KLD], KlB[2 * 32 * KLD];
  __shared__ __bf16 VhB[2 * 128 * VLD], VlB[2 * 128 * VLD];   // [d][key]
  __shared__ __bf16 Ph[8 * 16 * VLD], Pl[8 * 16 * VLD];

  const int tid   = threadIdx.x;
  const int lane  = tid & 31;
  const int wave  = tid >> 5;
  const int l16   = lane & 15;
  const int half8 = (lane < 16) ? 0 : 8;
  const int qtile = blockIdx.x;          // 0..15
  const int bh    = blockIdx.y;          // 0..63
  const int b = bh >> 4, h = bh & 15;
  const size_t bT = (size_t)b * T_;
  const float scale = 0.08838834764831845f;  // 1/sqrt(128)

  const __bf16* kbh = qkh + bT * QK_LD + DIM_ + h * HD_;  // K region base
  const __bf16* kbl = qkl + bT * QK_LD + DIM_ + h * HD_;
  const __bf16* vbh = vTh + (size_t)(b * H_ + h) * HD_ * T_;
  const __bf16* vbl = vTl + (size_t)(b * H_ + h) * HD_ * T_;

  // issue one step's K/V tile copies into LDS buffer `buf`
  auto issue_tiles = [&](int buf, int key0) {
    const int lh = lane >> 4, l3 = lane & 3, ld2 = lane >> 2;
#pragma unroll
    for (int k = 0; k < 2; ++k) {
      int row = wave * 4 + k * 2 + lh;                    // key within tile
      unsigned voff = (unsigned)((key0 + row) * (QK_LD * 2) + l16 * 16);
      async_cp16(&KhB[buf * 32 * KLD + row * KLD + l16 * 8], kbh, voff);
      async_cp16(&KlB[buf * 32 * KLD + row * KLD + l16 * 8], kbl, voff);
    }
#pragma unroll
    for (int k = 0; k < 2; ++k) {
      int d = wave * 16 + k * 8 + ld2;
      unsigned voff = (unsigned)(d * (T_ * 2) + key0 * 2 + l3 * 16);
      async_cp16(&VhB[buf * 128 * VLD + d * VLD + l3 * 8], vbh, voff);
      async_cp16(&VlB[buf * 128 * VLD + d * VLD + l3 * 8], vbl, voff);
    }
  };

  // Q fragments from the hi/lo planes (already split; pure loads)
  v16bf qh[4], ql[4];
  {
    const size_t qrow = bT + qtile * 128 + wave * 16 + l16;
    const __bf16* qb_h = qkh + qrow * QK_LD + h * HD_;
    const __bf16* qb_l = qkl + qrow * QK_LD + h * HD_;
#pragma unroll
    for (int c = 0; c < 4; ++c) {
      qh[c] = ldA(qb_h + c * 32, half8);
      ql[c] = ldA(qb_l + c * 32, half8);
    }
  }

  v8f o[8];
#pragma unroll
  for (int i = 0; i < 8; ++i) o[i] = v8f_zero();
  float mrow[8], srow[8];
#pragma unroll
  for (int i = 0; i < 8; ++i) { mrow[i] = -1e30f; srow[i] = 0.0f; }

  __bf16* Pwh = &Ph[wave * 16 * VLD];
  __bf16* Pwl = &Pl[wave * 16 * VLD];

  issue_tiles(0, 0);  // prologue

  for (int s = 0; s < T_ / 32; ++s) {
    wait_async0();       // own async copies for buffer s&1 done
    __syncthreads();     // everyone's copies visible; prev reads finished
    if (s + 1 < T_ / 32) issue_tiles((s + 1) & 1, (s + 1) * 32);

    const __bf16* Kch = &KhB[(s & 1) * 32 * KLD];
    const __bf16* Kcl = &KlB[(s & 1) * 32 * KLD];
    const __bf16* Vch = &VhB[(s & 1) * 128 * VLD];
    const __bf16* Vcl = &VlB[(s & 1) * 128 * VLD];

    // logits: 16 q x 32 keys
    v8f c0 = v8f_zero(), c1 = v8f_zero();
#pragma unroll
    for (int c = 0; c < 4; ++c) {
      const __bf16* k0h = &Kch[l16 * KLD + c * 32];
      const __bf16* k0l = &Kcl[l16 * KLD + c * 32];
      v16bf b0H = ldB(k0h, lane),            b0L = ldB(k0l, lane);
      v16bf b1H = ldB(k0h + 16 * KLD, lane), b1L = ldB(k0l + 16 * KLD, lane);
      c0 = wmma_bf16(qh[c], b0H, c0);
      c0 = wmma_bf16(qh[c], b0L, c0);
      c0 = wmma_bf16(ql[c], b0H, c0);
      c1 = wmma_bf16(qh[c], b1H, c1);
      c1 = wmma_bf16(qh[c], b1L, c1);
      c1 = wmma_bf16(ql[c], b1H, c1);
    }
#pragma unroll
    for (int i = 0; i < 8; ++i) { c0[i] *= scale; c1[i] *= scale; }

    // online softmax (row i lives on the 16 lanes of this half-wave)
    float alpha[8];
#pragma unroll
    for (int i = 0; i < 8; ++i) {
      float rm = hmax16(fmaxf(c0[i], c1[i]));
      float nm = fmaxf(mrow[i], rm);
      alpha[i] = __expf(mrow[i] - nm);
      mrow[i]  = nm;
      c0[i] = __expf(c0[i] - nm);
      c1[i] = __expf(c1[i] - nm);
      float rs = hsum16(c0[i] + c1[i]);
      srow[i] = srow[i] * alpha[i] + rs;
    }
#pragma unroll
    for (int nt = 0; nt < 8; ++nt)
#pragma unroll
      for (int i = 0; i < 8; ++i) o[nt][i] *= alpha[i];

    // P: C-layout -> A-layout via per-wave LDS (DS ops are in-order per wave)
#pragma unroll
    for (int i = 0; i < 8; ++i) {
      int prow = half8 + i;
      bfpair p0 = split1(c0[i]);
      bfpair p1 = split1(c1[i]);
      Pwh[prow * VLD + l16]      = p0.hi;
      Pwl[prow * VLD + l16]      = p0.lo;
      Pwh[prow * VLD + 16 + l16] = p1.hi;
      Pwl[prow * VLD + 16 + l16] = p1.lo;
    }
    asm volatile("" ::: "memory");
    v16bf pH = ldA(&Pwh[l16 * VLD], half8);
    v16bf pL = ldA(&Pwl[l16 * VLD], half8);

    // o += P @ V  (V^T in LDS: lane = d column, K = keys)
#pragma unroll
    for (int nt = 0; nt < 8; ++nt) {
      v16bf bH = ldB(&Vch[(nt * 16 + l16) * VLD], lane);
      v16bf bL = ldB(&Vcl[(nt * 16 + l16) * VLD], lane);
      o[nt] = wmma_bf16(pH, bH, o[nt]);
      o[nt] = wmma_bf16(pH, bL, o[nt]);
      o[nt] = wmma_bf16(pL, bH, o[nt]);
    }
  }

  // normalize and store y[b,t,h,d]
  float inv[8];
#pragma unroll
  for (int i = 0; i < 8; ++i) inv[i] = 1.0f / srow[i];
  const int m0 = qtile * 128 + wave * 16;
#pragma unroll
  for (int nt = 0; nt < 8; ++nt)
#pragma unroll
    for (int i = 0; i < 8; ++i) {
      int m = m0 + half8 + i;
      y[(bT + m) * DIM_ + h * HD_ + nt * 16 + l16] = o[nt][i] * inv[i];
    }
}

// ---------------------------------------------------------------------------
extern "C" void kernel_launch(void* const* d_in, const int* in_sizes, int n_in,
                              void* d_out, int out_size, void* d_ws, size_t ws_size,
                              hipStream_t stream) {
  const float* x      = (const float*)d_in[0];  // [B,T,DIM]
  const float* w_attn = (const float*)d_in[1];  // [3*DIM, DIM]
  const float* w_proj = (const float*)d_in[2];  // [DIM, DIM]
  float* out = (float*)d_out;                   // [B,T,DIM]

  char* w = (char*)d_ws;
  __bf16* qkh = (__bf16*)w;                       w += (size_t)8192 * QK_LD * 2;
  __bf16* qkl = (__bf16*)w;                       w += (size_t)8192 * QK_LD * 2;
  __bf16* vTh = (__bf16*)w;                       w += (size_t)B_ * H_ * HD_ * T_ * 2;
  __bf16* vTl = (__bf16*)w;                       w += (size_t)B_ * H_ * HD_ * T_ * 2;
  float*  y   = (float*)w;

  const int M = B_ * T_;  // 8192

  // 1) qkv = x @ w_attn^T  -> split planes (q|k row-major, V transposed)
  gemm_nt_kernel<<<dim3((3 * DIM_) / 128, M / 128), 256, 0, stream>>>(
      x, w_attn, nullptr, qkh, qkl, vTh, vTl, M, 3 * DIM_, DIM_, 1);

  // 2) RoPE on q|k planes in-place
  {
    int total = B_ * T_ * 2 * H_ * (HD_ / 2);  // 16,777,216
    rope_kernel<<<dim3(total / 256), 256, 0, stream>>>(qkh, qkl);
  }

  // 3) flash attention -> y (f32)
  flash_attn_kernel<<<dim3(T_ / 128, B_ * H_), 256, 0, stream>>>(
      qkh, qkl, vTh, vTl, y);

  // 4) out = y @ w_proj^T
  gemm_nt_kernel<<<dim3(DIM_ / 128, M / 128), 256, 0, stream>>>(
      y, w_proj, out, nullptr, nullptr, nullptr, nullptr, M, DIM_, DIM_, 0);
}
